// CausalSelfAttention_20306605376025
// MI455X (gfx1250) — compile-verified, matrix operand order checked by reading, not run on hardware
//
#include <hip/hip_runtime.h>

// ---------------------------------------------------------------------------
// CDNA5 (gfx1250) causal self-attention with RoPE + GQA, bf16 WMMA pipeline.
// All matmuls use v_wmma_f32_16x16x32_bf16 (wave32).
// GEMM uses double-buffered LDS tiles with GLOBAL_LOAD_ASYNC_TO_LDS_B128
// prefetch (ASYNCcnt-tracked) so the DMA for tile j+1 overlaps the WMMAs of
// tile j. Falls back to synchronous staging if the builtin is absent.
// ---------------------------------------------------------------------------

typedef __attribute__((ext_vector_type(16))) __bf16 v16bf;
typedef __attribute__((ext_vector_type(8)))  float  v8f;
typedef __attribute__((ext_vector_type(4)))  int    v4i;

union Frag16 { v16bf v; uint4 u[2]; __bf16 h[16]; };
union Chunk8 { uint4 u; __bf16 h[8]; };

#define LOG2E 1.44269504088896340736f

#if defined(__AMDGCN__) && __has_builtin(__builtin_amdgcn_global_load_async_to_lds_b128)
#define USE_ASYNC_LDS 1
typedef __attribute__((address_space(1))) v4i glob_v4i;
typedef __attribute__((address_space(3))) v4i lds_v4i;
#else
#define USE_ASYNC_LDS 0
#endif

// 16-byte global -> LDS copy (async when available)
__device__ __forceinline__ void copy16_g2l(const __bf16* g, __bf16* l) {
#if USE_ASYNC_LDS
    __builtin_amdgcn_global_load_async_to_lds_b128(
        (glob_v4i*)(__bf16*)g, (lds_v4i*)l, 0, 0);
#else
    *(uint4*)l = *(const uint4*)g;
#endif
}

__device__ __forceinline__ void async_lds_fence() {
#if USE_ASYNC_LDS
#if __has_builtin(__builtin_amdgcn_s_wait_asynccnt)
    __builtin_amdgcn_s_wait_asynccnt(0);
#else
    asm volatile("s_wait_asynccnt 0" ::: "memory");
#endif
#endif
}

// ---------------------------------------------------------------------------
// fp32 -> bf16 elementwise conversion
// ---------------------------------------------------------------------------
__global__ void f32_to_bf16(const float* __restrict__ in, __bf16* __restrict__ out, int n) {
    int i = blockIdx.x * blockDim.x + threadIdx.x;
    if (i < n) out[i] = (__bf16)in[i];
}

// ---------------------------------------------------------------------------
// RoPE in-place on [rows, nheads*128] bf16; row = b*T + t
// ---------------------------------------------------------------------------
__global__ void rope_kernel(__bf16* __restrict__ X, int rowsTotal, int nheads, int T) {
    int idx = blockIdx.x * blockDim.x + threadIdx.x;
    int total = rowsTotal * nheads * 64;
    if (idx >= total) return;
    int i   = idx & 63;
    int h   = (idx >> 6) % nheads;
    int row = idx / (nheads * 64);
    int t   = row % T;
    // inv_freq = 10000^(-2i/128) = exp2(-(i/64)*log2(10000))
    float invf = exp2f(-(float)i * (13.287712379549449f / 64.0f));
    float ang  = (float)t * invf;
    float c = __cosf(ang), s = __sinf(ang);
    size_t base = (size_t)row * (size_t)(nheads * 128) + (size_t)h * 128;
    float q1 = (float)X[base + i];
    float q2 = (float)X[base + i + 64];
    X[base + i]      = (__bf16)(q1 * c - q2 * s);
    X[base + i + 64] = (__bf16)(q2 * c + q1 * s);
}

// ---------------------------------------------------------------------------
// bf16 GEMM: C[M,N] = A[M,K] @ B[K,N], fp32 accumulate.
// Workgroup = 128 threads (4 waves), tile 128(M) x 64(N), K-step 32.
// Each wave computes 32x64 (2 M-subtiles x 4 N-subtiles): 8 WMMAs per K-step.
// Double-buffered LDS: tile j+1 is async-prefetched (A, 8KB/step) / staged
// (B transposed) while tile j's WMMAs run; s_wait_asynccnt at the top of the
// next iteration covers copies that had a full compute phase in flight.
// ---------------------------------------------------------------------------
template <bool OUT_F32>
__global__ __launch_bounds__(128) void gemm_bf16(const __bf16* __restrict__ A,
                                                 const __bf16* __restrict__ Bm,
                                                 void* __restrict__ Cv,
                                                 int M, int N, int K) {
    __shared__ __align__(16) __bf16 Al[2][128][32];
    __shared__ __align__(16) __bf16 Bt[2][64][32];

    const int tid   = threadIdx.x;
    const int lane  = tid & 31;
    const int wave  = tid >> 5;
    const int l16   = lane & 15;
    const int lhalf = lane >> 4;
    const int klane = lhalf * 8;
    const int mbase = blockIdx.x * 128;
    const int nbase = blockIdx.y * 64;

    v8f zero = {};
    v8f acc[2][4];
#pragma unroll
    for (int ms = 0; ms < 2; ++ms)
#pragma unroll
        for (int ns = 0; ns < 4; ++ns) acc[ms][ns] = zero;

    // Stage one K-tile (A async row-major, B synchronous transposed)
    auto stage = [&](int k0, int buf) {
        // A tile 128x32: 512 chunks of 8 bf16, async DMA to LDS
#pragma unroll
        for (int it = 0; it < 4; ++it) {
            int idx = it * 128 + tid;
            int row = idx >> 2;
            int ch  = idx & 3;
            copy16_g2l(A + (size_t)(mbase + row) * K + k0 + ch * 8,
                       &Al[buf][row][ch * 8]);
        }
        // B tile 32x64 transposed into Bt[buf][n][k]
#pragma unroll
        for (int it = 0; it < 2; ++it) {
            int idx = it * 128 + tid;
            int kr  = idx >> 3;
            int nc  = idx & 7;
            Chunk8 c;
            c.u = *(const uint4*)(Bm + (size_t)(k0 + kr) * N + nbase + nc * 8);
#pragma unroll
            for (int j = 0; j < 8; ++j) Bt[buf][nc * 8 + j][kr] = c.h[j];
        }
    };

    const int nk = K / 32;
    stage(0, 0);

    for (int j = 0; j < nk; ++j) {
        async_lds_fence();   // tile j's async copies complete (issued last iter)
        __syncthreads();     // staged data visible; prior compute done reading
        if (j + 1 < nk) stage((j + 1) * 32, (j + 1) & 1);  // prefetch next tile

        const int buf = j & 1;
        // A fragments: this wave's 32 rows (two 16-row subtiles)
        Frag16 af[2];
#pragma unroll
        for (int ms = 0; ms < 2; ++ms) {
            af[ms].u[0] = *(const uint4*)(&Al[buf][wave * 32 + ms * 16 + l16][klane]);
            af[ms].u[1] = *(const uint4*)(&Al[buf][wave * 32 + ms * 16 + l16][klane + 16]);
        }
#pragma unroll
        for (int ns = 0; ns < 4; ++ns) {
            Frag16 bf;
            bf.u[0] = *(const uint4*)(&Bt[buf][ns * 16 + l16][klane]);
            bf.u[1] = *(const uint4*)(&Bt[buf][ns * 16 + l16][klane + 16]);
#pragma unroll
            for (int ms = 0; ms < 2; ++ms)
                acc[ms][ns] = __builtin_amdgcn_wmma_f32_16x16x32_bf16(
                    false, af[ms].v, false, bf.v, (short)0, acc[ms][ns], false, false);
        }
    }

    // C/D layout: lanes 0-15 -> rows r (vgpr r), lanes 16-31 -> rows 8+r
#pragma unroll
    for (int ms = 0; ms < 2; ++ms) {
        const int mrow = mbase + wave * 32 + ms * 16 + (lhalf ? 8 : 0);
#pragma unroll
        for (int ns = 0; ns < 4; ++ns) {
            int col = nbase + ns * 16 + l16;
#pragma unroll
            for (int r = 0; r < 8; ++r) {
                float val = acc[ms][ns][r];
                if (OUT_F32)
                    ((float*)Cv)[(size_t)(mrow + r) * N + col] = val;
                else
                    ((__bf16*)Cv)[(size_t)(mrow + r) * N + col] = (__bf16)val;
            }
        }
    }
}

// ---------------------------------------------------------------------------
// Flash-style causal attention with GQA (H=16, Hkv=4, D=128).
// Grid: (B*H, T/64). Block: 128 threads (4 waves); each wave owns 16 q rows.
// Key blocks of 32; per block: 8 WMMAs for S = Q K^T, online softmax with
// __shfl_xor row reductions, P restaged via LDS into A-layout, 8 WMMAs for
// O += P V.
// ---------------------------------------------------------------------------
__global__ __launch_bounds__(128) void attn_kernel(const __bf16* __restrict__ Q,
                                                   const __bf16* __restrict__ Km,
                                                   const __bf16* __restrict__ Vm,
                                                   __bf16* __restrict__ Y, int T) {
    __shared__ __align__(16) __bf16 Kl[32][128];   // K block, row-major [key][d]
    __shared__ __align__(16) __bf16 Vt[128][32];   // V block transposed [d][key]
    __shared__ __align__(16) __bf16 Pl[4][16][32]; // per-wave P staging

    const int tid   = threadIdx.x;
    const int lane  = tid & 31;
    const int wave  = tid >> 5;
    const int l16   = lane & 15;
    const int lhalf = lane >> 4;
    const int klane = lhalf * 8;

    const int bh = blockIdx.x;          // b*16 + h
    const int b  = bh >> 4;
    const int h  = bh & 15;
    const int hk = h >> 2;              // GQA: 4 q heads per kv head
    const int qbase = blockIdx.y * 64;
    const int qrow  = qbase + wave * 16;

    // Q fragments: 4 d-chunks of 32, loaded straight from global (contiguous)
    Frag16 qf[4];
    {
        size_t qoff = (size_t)(b * T + qrow + l16) * 2048 + (size_t)h * 128;
#pragma unroll
        for (int c = 0; c < 4; ++c) {
            qf[c].u[0] = *(const uint4*)(Q + qoff + c * 32 + klane);
            qf[c].u[1] = *(const uint4*)(Q + qoff + c * 32 + klane + 16);
        }
    }

    v8f zero = {};
    v8f o[8];
#pragma unroll
    for (int i = 0; i < 8; ++i) o[i] = zero;
    float rowmax[8], rowsum[8];
#pragma unroll
    for (int r = 0; r < 8; ++r) { rowmax[r] = -3.0e38f; rowsum[r] = 0.0f; }
    const float scale = 0.08838834764831845f;  // 1/sqrt(128)

    const int nblocks = qbase / 32 + 2;  // covers keys up to qbase+63 (causal)
    for (int j = 0; j < nblocks; ++j) {
        const int kb = j * 32;
        __syncthreads();
        // Stage K (32x128) row-major (async) and V transposed (128x32)
#pragma unroll
        for (int it = 0; it < 4; ++it) {
            int idx = it * 128 + tid;   // 0..511
            int kr  = idx >> 4;         // 0..31
            int dc  = idx & 15;
            size_t goff = (size_t)(b * T + kb + kr) * 512 + (size_t)hk * 128 + dc * 8;
            copy16_g2l(Km + goff, &Kl[kr][dc * 8]);
            Chunk8 cv; cv.u = *(const uint4*)(Vm + goff);
#pragma unroll
            for (int jj = 0; jj < 8; ++jj) Vt[dc * 8 + jj][kr] = cv.h[jj];
        }
        async_lds_fence();
        __syncthreads();

        // S = Q K^T : two 16-key subtiles, contraction over D in 4 chunks
        v8f s01[2];
#pragma unroll
        for (int ss = 0; ss < 2; ++ss) {
            v8f sacc = zero;
#pragma unroll
            for (int c = 0; c < 4; ++c) {
                Frag16 kf;
                kf.u[0] = *(const uint4*)(&Kl[ss * 16 + l16][c * 32 + klane]);
                kf.u[1] = *(const uint4*)(&Kl[ss * 16 + l16][c * 32 + klane + 16]);
                sacc = __builtin_amdgcn_wmma_f32_16x16x32_bf16(
                    false, qf[c].v, false, kf.v, (short)0, sacc, false, false);
            }
            s01[ss] = sacc;
        }
        // scale + causal mask (element (m,n): m = r + 8*lhalf, n = lane%16)
#pragma unroll
        for (int ss = 0; ss < 2; ++ss) {
            int keyg = kb + ss * 16 + l16;
#pragma unroll
            for (int r = 0; r < 8; ++r) {
                int qg = qrow + r + lhalf * 8;
                float v = s01[ss][r] * scale;
                s01[ss][r] = (keyg <= qg) ? v : -3.0e38f;
            }
        }
        // online softmax: per-row reductions across the 16 lanes of each half
        float alpha[8];
#pragma unroll
        for (int r = 0; r < 8; ++r) {
            float mx = fmaxf(s01[0][r], s01[1][r]);
#pragma unroll
            for (int m = 1; m < 16; m <<= 1) mx = fmaxf(mx, __shfl_xor(mx, m, 32));
            float nm = fmaxf(rowmax[r], mx);
            alpha[r] = exp2f((rowmax[r] - nm) * LOG2E);
            rowmax[r] = nm;
            float p0 = exp2f((s01[0][r] - nm) * LOG2E);
            float p1 = exp2f((s01[1][r] - nm) * LOG2E);
            float ps = p0 + p1;
#pragma unroll
            for (int m = 1; m < 16; m <<= 1) ps += __shfl_xor(ps, m, 32);
            rowsum[r] = rowsum[r] * alpha[r] + ps;
            // stage P as bf16 into per-wave LDS (C-layout -> row-major)
            Pl[wave][r + lhalf * 8][l16]      = (__bf16)p0;
            Pl[wave][r + lhalf * 8][16 + l16] = (__bf16)p1;
        }
        // rescale O accumulators
#pragma unroll
        for (int ts = 0; ts < 8; ++ts)
#pragma unroll
            for (int r = 0; r < 8; ++r) o[ts][r] *= alpha[r];
        // reload P in WMMA A-layout
        Frag16 pf;
        pf.u[0] = *(const uint4*)(&Pl[wave][l16][klane]);
        pf.u[1] = *(const uint4*)(&Pl[wave][l16][klane + 16]);
        // O += P @ V (8 d-subtiles of 16)
#pragma unroll
        for (int ts = 0; ts < 8; ++ts) {
            Frag16 vf;
            vf.u[0] = *(const uint4*)(&Vt[ts * 16 + l16][klane]);
            vf.u[1] = *(const uint4*)(&Vt[ts * 16 + l16][klane + 16]);
            o[ts] = __builtin_amdgcn_wmma_f32_16x16x32_bf16(
                false, pf.v, false, vf.v, (short)0, o[ts], false, false);
        }
    }

    // normalize and write attention output (bf16, [B*T, H*D])
    size_t yrow0 = (size_t)(b * T + qrow + lhalf * 8);
#pragma unroll
    for (int ts = 0; ts < 8; ++ts) {
        int col = h * 128 + ts * 16 + l16;
#pragma unroll
        for (int r = 0; r < 8; ++r) {
            float val = o[ts][r] / rowsum[r];
            Y[(yrow0 + r) * 2048 + col] = (__bf16)val;
        }
    }
}

// ---------------------------------------------------------------------------
// Host launch
// ---------------------------------------------------------------------------
extern "C" void kernel_launch(void* const* d_in, const int* in_sizes, int n_in,
                              void* d_out, int out_size, void* d_ws, size_t ws_size,
                              hipStream_t stream) {
    (void)in_sizes; (void)n_in; (void)out_size; (void)ws_size;
    const float* x  = (const float*)d_in[0];
    const float* wq = (const float*)d_in[1];
    const float* wk = (const float*)d_in[2];
    const float* wv = (const float*)d_in[3];
    const float* wo = (const float*)d_in[4];

    constexpr int B = 2, T = 2048, C = 2048, Hq = 16, Hkv = 4, Dh = 128;
    constexpr int M = B * T;        // 4096
    constexpr int Nq = Hq * Dh;     // 2048
    constexpr int Nkv = Hkv * Dh;   // 512

    char* ws = (char*)d_ws;
    size_t off = 0;
    auto take = [&](size_t elems) -> __bf16* {
        __bf16* p = (__bf16*)(ws + off);
        off += (elems * sizeof(__bf16) + 255) & ~(size_t)255;
        return p;
    };
    __bf16* xb  = take((size_t)M * C);
    __bf16* wqb = take((size_t)C * Nq);
    __bf16* wkb = take((size_t)C * Nkv);
    __bf16* wvb = take((size_t)C * Nkv);
    __bf16* wob = take((size_t)Nq * C);
    __bf16* Qb  = take((size_t)M * Nq);
    __bf16* Kb  = take((size_t)M * Nkv);
    __bf16* Vb  = take((size_t)M * Nkv);
    __bf16* Yb  = take((size_t)M * Nq);

    auto cvt = [&](const float* src, __bf16* dst, int n) {
        f32_to_bf16<<<(n + 255) / 256, 256, 0, stream>>>(src, dst, n);
    };
    cvt(x, xb, M * C);
    cvt(wq, wqb, C * Nq);
    cvt(wk, wkb, C * Nkv);
    cvt(wv, wvb, C * Nkv);
    cvt(wo, wob, Nq * C);

    // QKV projections
    gemm_bf16<false><<<dim3(M / 128, Nq / 64), 128, 0, stream>>>(xb, wqb, Qb, M, Nq, C);
    gemm_bf16<false><<<dim3(M / 128, Nkv / 64), 128, 0, stream>>>(xb, wkb, Kb, M, Nkv, C);
    gemm_bf16<false><<<dim3(M / 128, Nkv / 64), 128, 0, stream>>>(xb, wvb, Vb, M, Nkv, C);

    // RoPE on Q and K
    rope_kernel<<<(M * Hq * 64) / 256, 256, 0, stream>>>(Qb, M, Hq, T);
    rope_kernel<<<(M * Hkv * 64) / 256, 256, 0, stream>>>(Kb, M, Hkv, T);

    // Causal GQA attention
    attn_kernel<<<dim3(B * Hq, T / 64), 128, 0, stream>>>(Qb, Kb, Vb, Yb, T);

    // Output projection (fp32 result into d_out)
    gemm_bf16<true><<<dim3(M / 128, C / 64), 128, 0, stream>>>(Yb, wob, d_out, M, C, Nq);
}
